// Model_38113539785439
// MI455X (gfx1250) — matrix operand order repaired and checked by reading, not compiled, not run on hardware
//
#include <hip/hip_runtime.h>

// ---------------- problem constants ----------------
#define BB   4
#define TT   2048
#define DD   2048
#define HH   16
#define KK   128          // head dim (K == V == 128)
#define HK   (HH*KK)      // 2048
#define NSTACK (5*HK)     // Wq|Wk|Wv|Wa|Wg stacked -> 10240
#define ROWS (BB*TT)      // 8192
#define EPS_G 1e-6f
#define LN_EPS 1e-5f

typedef __attribute__((ext_vector_type(16))) __bf16 v16bf;
typedef __attribute__((ext_vector_type(8)))  __bf16 v8bf;
typedef __attribute__((ext_vector_type(8)))  float  v8f;
typedef int v4i __attribute__((vector_size(4 * sizeof(int))));   // matches builtin param

#define GLOBAL_AS __attribute__((address_space(1)))
#define LDS_AS    __attribute__((address_space(3)))

#if defined(__has_builtin)
#  if __has_builtin(__builtin_amdgcn_global_load_async_to_lds_b128)
#    define HAVE_ASYNC_LDS 1
#  endif
#endif
#ifndef HAVE_ASYNC_LDS
#  define HAVE_ASYNC_LDS 0
#endif

// 16-byte global -> LDS copy; async (no VGPR round-trip) when available.
__device__ __forceinline__ void copy16_g2l(const __bf16* g, __bf16* l) {
#if HAVE_ASYNC_LDS
  __builtin_amdgcn_global_load_async_to_lds_b128(
      (GLOBAL_AS v4i*)g, (LDS_AS v4i*)l, /*offset=*/0, /*cpol=*/0);
#else
  *(v8bf*)l = *(const v8bf*)g;
#endif
}

__device__ __forceinline__ void wait_g2l() {
#if HAVE_ASYNC_LDS
#  if __has_builtin(__builtin_amdgcn_s_wait_asynccnt)
  __builtin_amdgcn_s_wait_asynccnt(0);
#  else
  asm volatile("s_wait_asynccnt 0x0" ::: "memory");
#  endif
#endif
}

// ---------------- fp32 -> bf16 convert ----------------
__global__ void f32_to_bf16(const float* __restrict__ src,
                            __bf16* __restrict__ dst, size_t n) {
  size_t i = (size_t)blockIdx.x * blockDim.x + threadIdx.x;
  size_t stride = (size_t)gridDim.x * blockDim.x;
  for (; i < n; i += stride) dst[i] = (__bf16)src[i];
}

// ---------------- bf16 WMMA GEMM:  C[M,N] = A[M,Kd] * B[N,Kd]^T ----------------
// A, B row-major with K contiguous. 128x128 block tile, K-step 32,
// double-buffered LDS (one barrier per K-step).
// 256 threads = 8 waves; each wave owns a 32(M) x 64(N) region = 2x4 WMMA tiles.
__global__ __launch_bounds__(256)
void gemm_bf16_nt(const __bf16* __restrict__ A, const __bf16* __restrict__ B,
                  float* __restrict__ C, int M, int N, int Kd) {
  __shared__ alignas(32) __bf16 As[2][128][32];
  __shared__ alignas(32) __bf16 Bs[2][128][32];

  const int tid  = threadIdx.x;
  const int lane = tid & 31;           // wave32
  const int wave = tid >> 5;
  const int wm   = wave & 3;           // 4 M-slices of 32 rows
  const int wn   = wave >> 2;          // 2 N-slices of 64 cols
  const int m0   = blockIdx.y * 128;
  const int n0   = blockIdx.x * 128;

  v8f acc[2][4] = {};                  // fp32 accumulators (8 VGPRs each)

  const int lrow = tid >> 1;           // 0..127
  const int lk   = (tid & 1) * 16;     // 0 or 16 halfs
  const __bf16* aG = A + (size_t)(m0 + lrow) * Kd + lk;
  const __bf16* bG = B + (size_t)(n0 + lrow) * Kd + lk;

  // fragment addressing (per ISA VGPR layouts)
  const int kb  = ((lane >> 4) & 1) * 8;   // A: lanes 0-15 K{0..7,16..23}, 16-31 K{8..15,24..31}
  const int kbB = ((lane >> 4) & 1) * 8;   // B: same K order as A (lanes 0-15 K{0..7,16..23}, 16-31 K{8..15,24..31})
  const int ml  = lane & 15;

  // preload tile 0
  copy16_g2l(aG,     &As[0][lrow][lk]);
  copy16_g2l(aG + 8, &As[0][lrow][lk + 8]);
  copy16_g2l(bG,     &Bs[0][lrow][lk]);
  copy16_g2l(bG + 8, &Bs[0][lrow][lk + 8]);
  wait_g2l();
  __syncthreads();

  int cur = 0;
  for (int k0 = 0; k0 < Kd; k0 += 32) {
    // stage next tile into the alternate buffer while computing on `cur`
    if (k0 + 32 < Kd) {
      const __bf16* ga = aG + k0 + 32;
      const __bf16* gb = bG + k0 + 32;
      __builtin_prefetch(ga + 32, 0, 1);        // global_prefetch_b8 one tile ahead
      __builtin_prefetch(gb + 32, 0, 1);
      copy16_g2l(ga,     &As[cur ^ 1][lrow][lk]);
      copy16_g2l(ga + 8, &As[cur ^ 1][lrow][lk + 8]);
      copy16_g2l(gb,     &Bs[cur ^ 1][lrow][lk]);
      copy16_g2l(gb + 8, &Bs[cur ^ 1][lrow][lk + 8]);
    }

    v16bf afrag[2], bfrag[4];
#pragma unroll
    for (int i = 0; i < 2; ++i) {
      const __bf16* ap = &As[cur][wm * 32 + i * 16 + ml][0];
      v8bf lo = *(const v8bf*)(ap + kb);
      v8bf hi = *(const v8bf*)(ap + 16 + kb);
      afrag[i] = __builtin_shufflevector(lo, hi, 0,1,2,3,4,5,6,7,8,9,10,11,12,13,14,15);
    }
#pragma unroll
    for (int j = 0; j < 4; ++j) {      // B frag: two 16B LDS loads per lane, same K order as A
      const __bf16* bp = &Bs[cur][wn * 64 + j * 16 + ml][0];
      v8bf lo = *(const v8bf*)(bp + kbB);
      v8bf hi = *(const v8bf*)(bp + 16 + kbB);
      bfrag[j] = __builtin_shufflevector(lo, hi, 0,1,2,3,4,5,6,7,8,9,10,11,12,13,14,15);
    }
#pragma unroll
    for (int i = 0; i < 2; ++i)
#pragma unroll
      for (int j = 0; j < 4; ++j)
        acc[i][j] = __builtin_amdgcn_wmma_f32_16x16x32_bf16(
            false, afrag[i], false, bfrag[j], (short)0, acc[i][j], false, false);

    wait_g2l();          // our async stores into buf[cur^1] done
    __syncthreads();     // everyone done writing next / reading cur
    cur ^= 1;
  }

  // C/D layout: lanes 0-15 -> M = r, lanes 16-31 -> M = 8+r; N = lane & 15
  const int mlane = lane & 15;
  const int mhi   = (lane >> 4) * 8;
#pragma unroll
  for (int i = 0; i < 2; ++i)
#pragma unroll
    for (int j = 0; j < 4; ++j) {
      const int n  = n0 + wn * 64 + j * 16 + mlane;
      const int mb = m0 + wm * 32 + i * 16 + mhi;
#pragma unroll
      for (int r = 0; r < 8; ++r)
        C[(size_t)(mb + r) * N + n] = acc[i][j][r];
    }
}

// ---------------- depthwise causal conv (K=4) + SiLU, relayout to (B,H,T,128) ----------------
__global__ void dconv_silu(const float* __restrict__ lin, int ld, int colOff,
                           const float* __restrict__ w, const float* __restrict__ bias,
                           float* __restrict__ dst) {
  size_t n = (size_t)ROWS * HK;
  size_t stride = (size_t)gridDim.x * blockDim.x;
  for (size_t idx = (size_t)blockIdx.x * blockDim.x + threadIdx.x; idx < n; idx += stride) {
    int c = (int)(idx % HK);
    size_t row = idx / HK;
    int t = (int)(row % TT);
    int b = (int)(row / TT);
    float acc = bias[c];
#pragma unroll
    for (int j = 0; j < 4; ++j) {
      int tt = t - 3 + j;
      if (tt >= 0)
        acc = fmaf(w[c * 4 + j], lin[(size_t)(b * TT + tt) * ld + colOff + c], acc);
    }
    float y = acc * (1.f / (1.f + __expf(-acc)));      // SiLU
    int h = c >> 7, d = c & 127;
    dst[(((size_t)(b * HH + h) * TT) + t) * KK + d] = y;
  }
}

// ---------------- a -> g = log(max(sigmoid(a_lin + ba), eps)), relayout ----------------
__global__ void gate_log_kernel(const float* __restrict__ proj, const float* __restrict__ ba,
                                float* __restrict__ gh) {
  size_t n = (size_t)ROWS * HK;
  size_t stride = (size_t)gridDim.x * blockDim.x;
  for (size_t idx = (size_t)blockIdx.x * blockDim.x + threadIdx.x; idx < n; idx += stride) {
    int c = (int)(idx % HK);
    size_t row = idx / HK;
    int t = (int)(row % TT);
    int b = (int)(row / TT);
    float a = proj[row * NSTACK + 3 * HK + c] + ba[c];
    a = 1.f / (1.f + __expf(-a));
    int h = c >> 7, d = c & 127;
    gh[(((size_t)(b * HH + h) * TT) + t) * KK + d] = __logf(fmaxf(a, EPS_G));
  }
}

// ---------------- beta = sigmoid(x @ Wb^T + bb) -> (B,H,T) ----------------
__global__ __launch_bounds__(256)
void beta_kernel(const float* __restrict__ x, const float* __restrict__ Wb,
                 const float* __restrict__ bb, float* __restrict__ beta) {
  __shared__ float xrow[DD];
  __shared__ float red[HH][16];
  const size_t row = blockIdx.x;            // b*T + t
  const int tid = threadIdx.x;
  for (int i = tid; i < DD; i += 256) xrow[i] = x[row * DD + i];
  __syncthreads();
  const int h = tid >> 4, l = tid & 15;
  float p = 0.f;
  for (int c = l; c < DD; c += 16) p = fmaf(xrow[c], Wb[(size_t)h * DD + c], p);
  red[h][l] = p;
  __syncthreads();
  if (l == 0) {
    float s = bb[h];
#pragma unroll
    for (int i = 0; i < 16; ++i) s += red[h][i];
    int b = (int)(row / TT), t = (int)(row % TT);
    beta[((size_t)(b * HH + h) * TT) + t] = 1.f / (1.f + __expf(-s));
  }
}

// ---------------- gated delta-rule scan; state in VGPRs (one v-column per lane) ----------------
#define CT 16
__global__ __launch_bounds__(128)
void kda_scan(const float* __restrict__ qh, const float* __restrict__ kh,
              const float* __restrict__ vh, const float* __restrict__ gh,
              const float* __restrict__ beta, float* __restrict__ oh) {
  __shared__ float sk[CT][KK];
  __shared__ float sq[CT][KK];
  __shared__ float sg[CT][KK];   // holds exp(g) after conversion
  __shared__ float sb[CT];
  const int bh  = blockIdx.x;    // 0..63
  const int tid = threadIdx.x;   // v column
  const float scale = 0.08838834764831845f;   // 128^-0.5
  const size_t base = (size_t)bh * TT;

  float S[KK];
#pragma unroll
  for (int i = 0; i < KK; ++i) S[i] = 0.f;

  for (int t0 = 0; t0 < TT; t0 += CT) {
    __syncthreads();
#pragma unroll
    for (int r = 0; r < CT; ++r) {
      size_t off = (base + t0 + r) * KK + tid;
      sk[r][tid] = kh[off];
      sq[r][tid] = qh[off];
      sg[r][tid] = gh[off];
    }
    if (tid < CT) sb[tid] = beta[base + t0 + tid];
#pragma unroll
    for (int r = 0; r < CT; ++r) sg[r][tid] = __expf(sg[r][tid]);  // own column: no race
    __syncthreads();

    for (int tt = 0; tt < CT; ++tt) {
      const float vt = vh[(base + t0 + tt) * KK + tid];
      float dot = 0.f;
#pragma unroll
      for (int k = 0; k < KK; ++k) {               // decay + k^T S
        S[k] *= sg[tt][k];
        dot = fmaf(sk[tt][k], S[k], dot);
      }
      const float be = sb[tt] * (vt - dot);        // beta * err
      float dq = 0.f;
#pragma unroll
      for (int k = 0; k < KK; ++k) {               // rank-1 update + q^T S
        S[k] = fmaf(be, sk[tt][k], S[k]);
        dq = fmaf(sq[tt][k], S[k], dq);
      }
      oh[(base + t0 + tt) * KK + tid] = scale * dq;
    }
  }
}

// ---------------- per-head LayerNorm + sigmoid gate -> bf16 y (B,T,H*V) ----------------
__global__ __launch_bounds__(128)
void ln_gate_kernel(const float* __restrict__ oh, const float* __restrict__ proj,
                    const float* __restrict__ ln_w, const float* __restrict__ ln_b,
                    __bf16* __restrict__ yb) {
  __shared__ float red[KK];
  const int bid = blockIdx.x;          // b*T*H + t*H + h
  const int h = bid % HH;
  const int t = (bid / HH) % TT;
  const int b = bid / (HH * TT);
  const int v = threadIdx.x;
  const float val = oh[(((size_t)(b * HH + h) * TT) + t) * KK + v];

  red[v] = val; __syncthreads();
  for (int s = 64; s > 0; s >>= 1) { if (v < s) red[v] += red[v + s]; __syncthreads(); }
  const float mu = red[0] * (1.f / KK);
  __syncthreads();
  const float d = val - mu;
  red[v] = d * d; __syncthreads();
  for (int s = 64; s > 0; s >>= 1) { if (v < s) red[v] += red[v + s]; __syncthreads(); }
  const float var = red[0] * (1.f / KK);

  float o = d * __frsqrt_rn(var + LN_EPS) * ln_w[v] + ln_b[v];
  const size_t row = (size_t)b * TT + t;
  const float gl = proj[row * NSTACK + 4 * HK + h * KK + v];      // Wg projection
  o *= 1.f / (1.f + __expf(-gl));
  yb[row * HK + h * KK + v] = (__bf16)o;
}

// ---------------- host-side orchestration ----------------
extern "C" void kernel_launch(void* const* d_in, const int* in_sizes, int n_in,
                              void* d_out, int out_size, void* d_ws, size_t ws_size,
                              hipStream_t stream) {
  const float* x    = (const float*)d_in[0];
  const float* Wq   = (const float*)d_in[1];
  const float* Wk   = (const float*)d_in[2];
  const float* Wv   = (const float*)d_in[3];
  const float* Wa   = (const float*)d_in[4];
  const float* ba   = (const float*)d_in[5];
  const float* Wb   = (const float*)d_in[6];
  const float* bb   = (const float*)d_in[7];
  const float* Wg   = (const float*)d_in[8];
  const float* Wo   = (const float*)d_in[9];
  const float* cqw  = (const float*)d_in[10];
  const float* cqb  = (const float*)d_in[11];
  const float* ckw  = (const float*)d_in[12];
  const float* ckb  = (const float*)d_in[13];
  const float* cvw  = (const float*)d_in[14];
  const float* cvb  = (const float*)d_in[15];
  const float* ln_w = (const float*)d_in[16];
  const float* ln_b = (const float*)d_in[17];
  float* out = (float*)d_out;

  char* ws = (char*)d_ws;
  size_t off = 0;
  auto alloc = [&](size_t bytes) -> void* {
    void* p = ws + off;
    off = (off + bytes + 255) & ~(size_t)255;
    return p;
  };
  __bf16* xb     = (__bf16*)alloc((size_t)ROWS * DD * 2);
  __bf16* wstack = (__bf16*)alloc((size_t)NSTACK * DD * 2);
  __bf16* wob    = (__bf16*)alloc((size_t)DD * HK * 2);
  float*  proj   = (float*)alloc((size_t)ROWS * NSTACK * 4);
  float*  qh     = (float*)alloc((size_t)ROWS * HK * 4);
  float*  kh     = (float*)alloc((size_t)ROWS * HK * 4);
  float*  vh     = (float*)alloc((size_t)ROWS * HK * 4);
  float*  gh     = (float*)alloc((size_t)ROWS * HK * 4);
  float*  beta   = (float*)alloc((size_t)BB * HH * TT * 4);
  float*  oh     = (float*)alloc((size_t)ROWS * HK * 4);
  __bf16* yb     = (__bf16*)alloc((size_t)ROWS * HK * 2);
  (void)ws_size; (void)in_sizes; (void)n_in; (void)out_size;

  const size_t nW = (size_t)HK * DD;
  // convert activations + weights to bf16
  f32_to_bf16<<<4096, 256, 0, stream>>>(x,  xb, (size_t)ROWS * DD);
  f32_to_bf16<<<2048, 256, 0, stream>>>(Wq, wstack + 0 * nW, nW);
  f32_to_bf16<<<2048, 256, 0, stream>>>(Wk, wstack + 1 * nW, nW);
  f32_to_bf16<<<2048, 256, 0, stream>>>(Wv, wstack + 2 * nW, nW);
  f32_to_bf16<<<2048, 256, 0, stream>>>(Wa, wstack + 3 * nW, nW);
  f32_to_bf16<<<2048, 256, 0, stream>>>(Wg, wstack + 4 * nW, nW);
  f32_to_bf16<<<2048, 256, 0, stream>>>(Wo, wob, nW);

  // fused projection GEMM: (8192 x 2048) x (10240 x 2048)^T
  gemm_bf16_nt<<<dim3(NSTACK / 128, ROWS / 128), 256, 0, stream>>>(
      xb, wstack, proj, ROWS, NSTACK, DD);

  // depthwise causal conv + SiLU for q/k/v (relayout to (B,H,T,128))
  dconv_silu<<<8192, 256, 0, stream>>>(proj, NSTACK, 0 * HK, cqw, cqb, qh);
  dconv_silu<<<8192, 256, 0, stream>>>(proj, NSTACK, 1 * HK, ckw, ckb, kh);
  dconv_silu<<<8192, 256, 0, stream>>>(proj, NSTACK, 2 * HK, cvw, cvb, vh);

  // channel log-gate and beta
  gate_log_kernel<<<8192, 256, 0, stream>>>(proj, ba, gh);
  beta_kernel<<<ROWS, 256, 0, stream>>>(x, Wb, bb, beta);

  // sequential gated delta-rule scan, state in registers
  kda_scan<<<BB * HH, 128, 0, stream>>>(qh, kh, vh, gh, beta, oh);

  // per-head LayerNorm + output gate -> bf16
  ln_gate_kernel<<<ROWS * HH, 128, 0, stream>>>(oh, proj, ln_w, ln_b, yb);

  // final GEMM: (8192 x 2048) x (2048 x 2048)^T -> d_out
  gemm_bf16_nt<<<dim3(HK / 128, ROWS / 128), 256, 0, stream>>>(
      yb, wob, out, ROWS, HK, DD);
}
